// GAT2_6631429505167
// MI455X (gfx1250) — matrix operand order repaired and checked by reading, not compiled, not run on hardware
//
#include <hip/hip_runtime.h>

// ---------- helpers ----------
__device__ __forceinline__ unsigned fmax_key(float f) {
  unsigned b = __float_as_uint(f);
  return (b & 0x80000000u) ? ~b : (b | 0x80000000u);
}
__device__ __forceinline__ float fmax_decode(unsigned k) {
  unsigned b = (k & 0x80000000u) ? (k ^ 0x80000000u) : ~k;
  return __uint_as_float(b);
}
__device__ __forceinline__ float lrelu(float x) { return x < 0.f ? 0.2f * x : x; }
__device__ __forceinline__ float elu1(float x) { return x > 0.f ? x : (__expf(x) - 1.f); }

// ---------- c1[h] = dot(Wt1_h, a_tp1_h), c2 = dot(Wt2, a_tp2) ----------
__global__ void prep_c(const float* __restrict__ Wt1, const float* __restrict__ atp1,
                       const float* __restrict__ Wt2, const float* __restrict__ atp2,
                       float* __restrict__ c1, float* __restrict__ c2) {
  int t = threadIdx.x;
  if (t < 8) {
    float s = 0.f;
    for (int f = 0; f < 16; ++f) s += Wt1[t * 16 + f] * atp1[t * 16 + f];
    c1[t] = s;
  }
  if (t == 8) {
    float s = 0.f;
    for (int f = 0; f < 128; ++f) s += Wt2[f] * atp2[f];
    *c2 = s;
  }
}

// ---------- C[M,128] = A[M,128] * B[128,128], bf16 WMMA, fp32 accum ----------
// One wave per 16-row strip; 4 K-steps of v_wmma_f32_16x16x32_bf16 per 16x16 tile,
// 8 column tiles per wave (A fragments loaded once, reused 8x).
__global__ void __launch_bounds__(256)
wmma_gemm_k128(const float* __restrict__ A, const float* __restrict__ B,
               float* __restrict__ C, int M) {
  typedef __attribute__((ext_vector_type(16))) __bf16 v16bf;
  typedef __attribute__((ext_vector_type(8))) float v8f;

  const int lane = threadIdx.x & 31;
  const int strip = blockIdx.x * (blockDim.x >> 5) + (threadIdx.x >> 5);
  const int m0 = strip * 16;
  if (m0 >= M) return;

  const int half = lane >> 4;   // 0: lanes 0-15, 1: lanes 16-31
  const int mr = lane & 15;
  const float* Arow = A + (size_t)(m0 + mr) * 128;

  // A fragment layout (16-bit A 16x32, ISA 7.12.2):
  // lane<16: VGPR g in 0..3 -> K=2g,2g+1 ; g in 4..7 -> K=16+2(g-4)
  // lane>=16: same +8
  v16bf afrag[4];
#pragma unroll
  for (int ks = 0; ks < 4; ++ks) {
#pragma unroll
    for (int g = 0; g < 8; ++g) {
      const int kb = ((g < 4) ? 0 : 16) + 8 * half + 2 * (g & 3);
      const float2 f2 = *(const float2*)(Arow + ks * 32 + kb);
      afrag[ks][2 * g] = (__bf16)f2.x;
      afrag[ks][2 * g + 1] = (__bf16)f2.y;
    }
  }

  const int nh = lane & 15;
  const int koff = half * 16;
  const int mh = half * 8;

#pragma unroll
  for (int nt = 0; nt < 8; ++nt) {
    const int n0 = nt * 16;
    v8f acc = {};
#pragma unroll
    for (int ks = 0; ks < 4; ++ks) {
      // B fragment (32x16): N=lane&15; VGPR v in 0..3 -> K=2v,2v+1 ; v in 4..7 -> K=16+2(v-4) ; lane>=16 -> K+8
      v16bf bfrag;
#pragma unroll
      for (int v = 0; v < 8; ++v) {
        const int k = ks * 32 + ((v < 4) ? 0 : 16) + 8 * half + 2 * (v & 3);
        bfrag[2 * v] = (__bf16)B[(size_t)k * 128 + n0 + nh];
        bfrag[2 * v + 1] = (__bf16)B[(size_t)(k + 1) * 128 + n0 + nh];
      }
      acc = __builtin_amdgcn_wmma_f32_16x16x32_bf16(
          false, afrag[ks], false, bfrag, (short)0, acc, false, false);
    }
    // C/D layout: VGPR r -> M = mh + r, N = nh
#pragma unroll
    for (int r = 0; r < 8; ++r)
      C[(size_t)(m0 + mh + r) * 128 + n0 + nh] = acc[r];
  }
}

// ---------- per-node attention score dots ----------
__global__ void node_scores_h8(const float* __restrict__ proj, const float* __restrict__ a_src,
                               const float* __restrict__ a_trg, float* __restrict__ ssrc,
                               float* __restrict__ strg, int N) {
  int i = blockIdx.x * blockDim.x + threadIdx.x;   // i = n*8 + h
  if (i >= N * 8) return;
  int h = i & 7;
  const float* p = proj + (size_t)(i >> 3) * 128 + h * 16;
  const float* as = a_src + h * 16;
  const float* at = a_trg + h * 16;
  float ss = 0.f, st = 0.f;
#pragma unroll
  for (int f = 0; f < 16; ++f) { float v = p[f]; ss += v * as[f]; st += v * at[f]; }
  ssrc[i] = ss;
  strg[i] = st;
}

__global__ void node_scores_h1(const float* __restrict__ proj, const float* __restrict__ a_src,
                               const float* __restrict__ a_trg, float* __restrict__ ssrc,
                               float* __restrict__ strg, int N) {
  int n = blockIdx.x * blockDim.x + threadIdx.x;
  if (n >= N) return;
  const float* p = proj + (size_t)n * 128;
  float ss = 0.f, st = 0.f;
  for (int f = 0; f < 128; f += 4) {
    float4 pv = *(const float4*)(p + f);
    float4 as = *(const float4*)(a_src + f);
    float4 at = *(const float4*)(a_trg + f);
    ss += pv.x * as.x + pv.y * as.y + pv.z * as.z + pv.w * as.w;
    st += pv.x * at.x + pv.y * at.y + pv.z * at.z + pv.w * at.w;
  }
  ssrc[n] = ss;
  strg[n] = st;
}

// ---------- global max over all edge scores (float via ordered-uint atomicMax) ----------
__global__ void edge_max_h8(const int* __restrict__ ei, const float* __restrict__ ep,
                            const float* __restrict__ ssrc, const float* __restrict__ strg,
                            const float* __restrict__ c1, unsigned* __restrict__ gmax, int E) {
  int e = blockIdx.x * blockDim.x + threadIdx.x;
  float m = -3.0e38f;
  if (e < E) {
    int s = ei[e], t = ei[E + e];
    float p = ep[e];
#pragma unroll
    for (int h = 0; h < 8; ++h)
      m = fmaxf(m, lrelu(ssrc[s * 8 + h] + strg[t * 8 + h] + p * c1[h]));
  }
#pragma unroll
  for (int off = 16; off > 0; off >>= 1)
    m = fmaxf(m, __shfl_xor(m, off, 32));
  if ((threadIdx.x & 31) == 0) atomicMax(gmax, fmax_key(m));
}

__global__ void edge_max_h1(const int* __restrict__ ei, const float* __restrict__ ep,
                            const float* __restrict__ ssrc, const float* __restrict__ strg,
                            const float* __restrict__ c2, unsigned* __restrict__ gmax, int E) {
  int e = blockIdx.x * blockDim.x + threadIdx.x;
  float m = -3.0e38f;
  if (e < E) {
    int s = ei[e], t = ei[E + e];
    m = lrelu(ssrc[s] + strg[t] + ep[e] * c2[0]);
  }
#pragma unroll
  for (int off = 16; off > 0; off >>= 1)
    m = fmaxf(m, __shfl_xor(m, off, 32));
  if ((threadIdx.x & 31) == 0) atomicMax(gmax, fmax_key(m));
}

// ---------- denom = segment_sum(exp(score - gmax), trg) ----------
__global__ void edge_denom_h8(const int* __restrict__ ei, const float* __restrict__ ep,
                              const float* __restrict__ ssrc, const float* __restrict__ strg,
                              const float* __restrict__ c1, const unsigned* __restrict__ gmax,
                              float* __restrict__ denom, int E) {
  int e = blockIdx.x * blockDim.x + threadIdx.x;
  if (e >= E) return;
  float gm = fmax_decode(*gmax);
  int s = ei[e], t = ei[E + e];
  float p = ep[e];
#pragma unroll
  for (int h = 0; h < 8; ++h) {
    float sc = lrelu(ssrc[s * 8 + h] + strg[t * 8 + h] + p * c1[h]);
    atomicAdd(&denom[t * 8 + h], __expf(sc - gm));
  }
}

__global__ void edge_denom_h1(const int* __restrict__ ei, const float* __restrict__ ep,
                              const float* __restrict__ ssrc, const float* __restrict__ strg,
                              const float* __restrict__ c2, const unsigned* __restrict__ gmax,
                              float* __restrict__ denom, int E) {
  int e = blockIdx.x * blockDim.x + threadIdx.x;
  if (e >= E) return;
  float gm = fmax_decode(*gmax);
  int s = ei[e], t = ei[E + e];
  float sc = lrelu(ssrc[s] + strg[t] + ep[e] * c2[0]);
  atomicAdd(&denom[t], __expf(sc - gm));
}

// ---------- out[trg] += proj[src] * att  (one wave per edge, 4 feats/lane) ----------
__global__ void edge_agg_h8(const int* __restrict__ ei, const float* __restrict__ ep,
                            const float* __restrict__ ssrc, const float* __restrict__ strg,
                            const float* __restrict__ c1, const float* __restrict__ denom,
                            const unsigned* __restrict__ gmax, const float* __restrict__ proj,
                            float* __restrict__ out, int E) {
  int tid = blockIdx.x * blockDim.x + threadIdx.x;
  int e = tid >> 5;
  if (e >= E) return;
  int lane = tid & 31;
  int s = ei[e], t = ei[E + e];
  float p = ep[e];
  int h = lane >> 2;  // 16 features per head -> 4 lanes per head
  float sc = lrelu(ssrc[s * 8 + h] + strg[t * 8 + h] + p * c1[h]);
  float att = __expf(sc - fmax_decode(*gmax)) / (denom[t * 8 + h] + 1e-16f);
  float4 pv = *(const float4*)(proj + (size_t)s * 128 + lane * 4);
  float* o = out + (size_t)t * 128 + lane * 4;
  atomicAdd(o + 0, pv.x * att);
  atomicAdd(o + 1, pv.y * att);
  atomicAdd(o + 2, pv.z * att);
  atomicAdd(o + 3, pv.w * att);
}

__global__ void edge_agg_h1(const int* __restrict__ ei, const float* __restrict__ ep,
                            const float* __restrict__ ssrc, const float* __restrict__ strg,
                            const float* __restrict__ c2, const float* __restrict__ denom,
                            const unsigned* __restrict__ gmax, const float* __restrict__ proj,
                            float* __restrict__ out, int E) {
  int tid = blockIdx.x * blockDim.x + threadIdx.x;
  int e = tid >> 5;
  if (e >= E) return;
  int lane = tid & 31;
  int s = ei[e], t = ei[E + e];
  float sc = lrelu(ssrc[s] + strg[t] + ep[e] * c2[0]);
  float att = __expf(sc - fmax_decode(*gmax)) / (denom[t] + 1e-16f);
  float4 pv = *(const float4*)(proj + (size_t)s * 128 + lane * 4);
  float* o = out + (size_t)t * 128 + lane * 4;
  atomicAdd(o + 0, pv.x * att);
  atomicAdd(o + 1, pv.y * att);
  atomicAdd(o + 2, pv.z * att);
  atomicAdd(o + 3, pv.w * att);
}

// ---------- layer finalizers ----------
__global__ void finalize_l1(const float* __restrict__ out1, const float* __restrict__ skip1,
                            const float* __restrict__ b, float* __restrict__ h1v, int NF) {
  int i = blockIdx.x * blockDim.x + threadIdx.x;
  if (i >= NF) return;
  h1v[i] = elu1(out1[i] + skip1[i] + b[i & 127]);
}

__global__ void finalize_l2(float* __restrict__ dout, const float* __restrict__ h1v,
                            const float* __restrict__ b, int NF) {
  int i = blockIdx.x * blockDim.x + threadIdx.x;
  if (i >= NF) return;
  dout[i] = elu1(dout[i] + h1v[i] + b[i & 127]);  // Wskip=None -> identity skip; H=1 mean = identity
}

// ---------- driver ----------
extern "C" void kernel_launch(void* const* d_in, const int* in_sizes, int n_in,
                              void* d_out, int out_size, void* d_ws, size_t ws_size,
                              hipStream_t stream) {
  const float* x    = (const float*)d_in[0];
  const int*   ei   = (const int*)d_in[1];
  const float* ep   = (const float*)d_in[2];
  const float* W1   = (const float*)d_in[3];
  const float* Wt1  = (const float*)d_in[4];
  const float* as1  = (const float*)d_in[5];
  const float* at1  = (const float*)d_in[6];
  const float* atp1 = (const float*)d_in[7];
  const float* b1   = (const float*)d_in[8];
  const float* Wsk  = (const float*)d_in[9];
  const float* W2   = (const float*)d_in[10];
  const float* Wt2  = (const float*)d_in[11];
  const float* as2  = (const float*)d_in[12];
  const float* at2  = (const float*)d_in[13];
  const float* atp2 = (const float*)d_in[14];
  const float* b2   = (const float*)d_in[15];

  const int N = in_sizes[0] / 128;
  const int E = in_sizes[2];
  const size_t NF = (size_t)N * 128;

  float* ws    = (float*)d_ws;
  float* proj1 = ws;
  float* skip1 = proj1 + NF;
  float* h1v   = skip1 + NF;
  float* proj2 = h1v + NF;
  float* out1  = proj2 + NF;
  float* ssrc1 = out1 + NF;
  float* strg1 = ssrc1 + (size_t)N * 8;
  float* den1  = strg1 + (size_t)N * 8;
  float* ssrc2 = den1 + (size_t)N * 8;
  float* strg2 = ssrc2 + N;
  float* den2  = strg2 + N;
  float* c1    = den2 + N;
  float* c2    = c1 + 8;
  unsigned* gmax1 = (unsigned*)(c2 + 1);
  float* dout = (float*)d_out;

  // zero accumulators (graph-capture-safe memset nodes); gmax=0 == "-inf" key
  hipMemsetAsync(out1, 0, NF * sizeof(float), stream);
  hipMemsetAsync(den1, 0, (size_t)N * 8 * sizeof(float), stream);
  hipMemsetAsync(den2, 0, (size_t)N * sizeof(float), stream);
  hipMemsetAsync(gmax1, 0, 2 * sizeof(unsigned), stream);
  hipMemsetAsync(dout, 0, NF * sizeof(float), stream);

  prep_c<<<1, 32, 0, stream>>>(Wt1, atp1, Wt2, atp2, c1, c2);

  const int strips = (N + 15) / 16;
  const int gblocks = (strips + 7) / 8;   // 8 waves/block
  const int eb = (E + 255) / 256;
  const int eb32 = (int)(((long long)E * 32 + 255) / 256);
  const int nfb = (int)((NF + 255) / 256);

  // ---- layer 1 ----
  wmma_gemm_k128<<<gblocks, 256, 0, stream>>>(x, W1, proj1, N);
  wmma_gemm_k128<<<gblocks, 256, 0, stream>>>(x, Wsk, skip1, N);
  node_scores_h8<<<(N * 8 + 255) / 256, 256, 0, stream>>>(proj1, as1, at1, ssrc1, strg1, N);
  edge_max_h8<<<eb, 256, 0, stream>>>(ei, ep, ssrc1, strg1, c1, gmax1, E);
  edge_denom_h8<<<eb, 256, 0, stream>>>(ei, ep, ssrc1, strg1, c1, gmax1, den1, E);
  edge_agg_h8<<<eb32, 256, 0, stream>>>(ei, ep, ssrc1, strg1, c1, den1, gmax1, proj1, out1, E);
  finalize_l1<<<nfb, 256, 0, stream>>>(out1, skip1, b1, h1v, (int)NF);

  // ---- layer 2 ----
  wmma_gemm_k128<<<gblocks, 256, 0, stream>>>(h1v, W2, proj2, N);
  node_scores_h1<<<(N + 255) / 256, 256, 0, stream>>>(proj2, as2, at2, ssrc2, strg2, N);
  edge_max_h1<<<eb, 256, 0, stream>>>(ei, ep, ssrc2, strg2, c2, gmax1 + 1, E);
  edge_denom_h1<<<eb, 256, 0, stream>>>(ei, ep, ssrc2, strg2, c2, gmax1 + 1, den2, E);
  edge_agg_h1<<<eb32, 256, 0, stream>>>(ei, ep, ssrc2, strg2, c2, den2, gmax1 + 1, proj2, dout, E);
  finalize_l2<<<nfb, 256, 0, stream>>>(dout, h1v, b2, (int)NF);
}
